// TransformerBlock_20830591386391
// MI455X (gfx1250) — compile-verified, matrix operand order checked by reading, not run on hardware
//
#include <hip/hip_runtime.h>
#include <math.h>

// ---------------- Types for CDNA5 WMMA (gfx1250, wave32) ----------------
typedef __bf16 bf16_t;
typedef __attribute__((ext_vector_type(16))) __bf16 v16bf;
typedef __attribute__((ext_vector_type(8)))  __bf16 v8bf;
typedef __attribute__((ext_vector_type(4)))  __bf16 v4bf;
typedef __attribute__((ext_vector_type(8)))  float  v8f;

#define TPAD 40            // LDS row stride (bf16): 80B -> b64/b128 aligned, spreads banks
#define MAXSLOTS 8704      // padded MoE slot capacity, multiple of 128 (2N=6400 + 16*127 <= 8432)

static __device__ __forceinline__ v16bf make_frag(const bf16_t* lo, const bf16_t* hi) {
    v8bf a = *(const v8bf*)lo;
    v8bf b = *(const v8bf*)hi;
    return __builtin_shufflevector(a, b, 0,1,2,3,4,5,6,7,8,9,10,11,12,13,14,15);
}

static __device__ __forceinline__ v4bf pack4(float4 v) {
    v4bf p;
    p[0] = (bf16_t)v.x; p[1] = (bf16_t)v.y; p[2] = (bf16_t)v.z; p[3] = (bf16_t)v.w;
    return p;
}

// ---------------- 128x64 block GEMM, bf16 WMMA, f32 accumulate, double-buffered regs ----------
// Requirements (all call sites satisfy): M%128==0, N%64==0, K%32==0, 16B-aligned rows.
// rowmap: if non-null, A row index = rowmap[m] (-1 => zero row). C row index is always m.
static __device__ __forceinline__ void gemm_block(
    const float* __restrict__ A, int lda, const int* __restrict__ rowmap,
    const float* __restrict__ B, int ldb,
    float* __restrict__ C, int ldc,
    const float* __restrict__ bias, const float* __restrict__ residual,
    int K, int relu, int row_base, int col_base,
    bf16_t (*As)[TPAD], bf16_t (*Bs)[TPAD])
{
    const int tid  = threadIdx.x;
    const int lane = tid & 31;
    const int wave = tid >> 5;      // 0..7
    const int wm   = wave & 3;      // 32-row band within 128
    const int wn   = wave >> 2;     // 32-col band within 64
    const int g    = lane >> 4;     // half-wave
    const int lr   = lane & 15;

    v8f acc00 = {}, acc01 = {}, acc10 = {}, acc11 = {};

    float4 arg[4], brg[2];

    // prologue loads for k0 = 0 (branchless rowmap gather, b128 global loads)
    #pragma unroll
    for (int i = 0; i < 4; ++i) {
        int id = i * 256 + tid;            // A tile: 128x32 = 1024 quads
        int r = id >> 3, kq = id & 7;
        int gm  = row_base + r;
        int src = rowmap ? rowmap[gm] : gm;
        int sc  = src < 0 ? 0 : src;
        float4 v = *(const float4*)(A + (size_t)sc * lda + kq * 4);
        if (src < 0) { v.x = 0.f; v.y = 0.f; v.z = 0.f; v.w = 0.f; }
        arg[i] = v;
    }
    #pragma unroll
    for (int i = 0; i < 2; ++i) {
        int id = i * 256 + tid;            // B tile: 32x64 = 512 quads
        int kk = id >> 4, nq = id & 15;
        brg[i] = *(const float4*)(B + (size_t)kk * ldb + col_base + nq * 4);
    }

    for (int k0 = 0; k0 < K; k0 += 32) {
        __syncthreads();                   // previous iteration's readers done
        // commit staged registers to LDS as bf16
        #pragma unroll
        for (int i = 0; i < 4; ++i) {
            int id = i * 256 + tid;
            int r = id >> 3, kq = id & 7;
            *(v4bf*)&As[r][kq * 4] = pack4(arg[i]);
        }
        #pragma unroll
        for (int i = 0; i < 2; ++i) {
            int id = i * 256 + tid;
            int kk = id >> 4, nq = id & 15;
            v4bf p = pack4(brg[i]);
            Bs[nq * 4 + 0][kk] = p[0];
            Bs[nq * 4 + 1][kk] = p[1];
            Bs[nq * 4 + 2][kk] = p[2];
            Bs[nq * 4 + 3][kk] = p[3];
        }
        // issue next tile's global loads (overlap with WMMA below)
        if (k0 + 32 < K) {
            int k1 = k0 + 32;
            #pragma unroll
            for (int i = 0; i < 4; ++i) {
                int id = i * 256 + tid;
                int r = id >> 3, kq = id & 7;
                int gm  = row_base + r;
                int src = rowmap ? rowmap[gm] : gm;
                int sc  = src < 0 ? 0 : src;
                float4 v = *(const float4*)(A + (size_t)sc * lda + k1 + kq * 4);
                if (src < 0) { v.x = 0.f; v.y = 0.f; v.z = 0.f; v.w = 0.f; }
                arg[i] = v;
            }
            #pragma unroll
            for (int i = 0; i < 2; ++i) {
                int id = i * 256 + tid;
                int kk = id >> 4, nq = id & 15;
                brg[i] = *(const float4*)(B + (size_t)(k1 + kk) * ldb + col_base + nq * 4);
            }
        }
        __syncthreads();                   // LDS tile visible

        const int ar0 = wm * 32 + lr;
        const int ar1 = ar0 + 16;
        v16bf a0 = make_frag(&As[ar0][g * 8], &As[ar0][16 + g * 8]);
        v16bf a1 = make_frag(&As[ar1][g * 8], &As[ar1][16 + g * 8]);
        const int bn0 = wn * 32 + lr;
        v16bf b0 = make_frag(&Bs[bn0][g * 8],      &Bs[bn0][16 + g * 8]);
        v16bf b1 = make_frag(&Bs[bn0 + 16][g * 8], &Bs[bn0 + 16][16 + g * 8]);

        acc00 = __builtin_amdgcn_wmma_f32_16x16x32_bf16(false, a0, false, b0, (short)0, acc00, false, false);
        acc01 = __builtin_amdgcn_wmma_f32_16x16x32_bf16(false, a0, false, b1, (short)0, acc01, false, false);
        acc10 = __builtin_amdgcn_wmma_f32_16x16x32_bf16(false, a1, false, b0, (short)0, acc10, false, false);
        acc11 = __builtin_amdgcn_wmma_f32_16x16x32_bf16(false, a1, false, b1, (short)0, acc11, false, false);
    }

    // epilogue: C-layout lane l: VGPR r -> row r+8*(l/16), col l%16
    #pragma unroll
    for (int ai = 0; ai < 2; ++ai) {
        #pragma unroll
        for (int r = 0; r < 8; ++r) {
            int m = row_base + wm * 32 + ai * 16 + r + 8 * g;
            #pragma unroll
            for (int f = 0; f < 2; ++f) {
                int n = col_base + wn * 32 + f * 16 + lr;
                float v;
                if (ai == 0) v = (f == 0) ? acc00[r] : acc01[r];
                else         v = (f == 0) ? acc10[r] : acc11[r];
                if (bias)     v += bias[n];
                if (residual) v += residual[(size_t)m * ldc + n];
                if (relu)     v = fmaxf(v, 0.0f);
                C[(size_t)m * ldc + n] = v;
            }
        }
    }
}

__global__ void __launch_bounds__(256) gemm_kernel(
    const float* A, int lda, const float* B, int ldb, float* C, int ldc,
    const float* bias, const float* residual, int K, int relu)
{
    __shared__ bf16_t As[128][TPAD];
    __shared__ bf16_t Bs[64][TPAD];
    gemm_block(A, lda, nullptr, B, ldb, C, ldc, bias, residual,
               K, relu, blockIdx.y * 128, blockIdx.x * 64, As, Bs);
}

// ---------------- MoE expert GEMMs (tile-uniform expert via 128-aligned slot bases) ----------
__global__ void __launch_bounds__(256) moe_ffn1_kernel(
    const float* h2, const float* w1, const float* b1, float* hidden,
    const int* token_of_slot, const int* expert_of_slot)
{
    int row_base = blockIdx.y * 128;
    int e = expert_of_slot[row_base];
    if (e < 0) return;
    __shared__ bf16_t As[128][TPAD];
    __shared__ bf16_t Bs[64][TPAD];
    gemm_block(h2, 768, token_of_slot, w1 + (size_t)e * 768 * 3072, 3072,
               hidden, 3072, b1 + (size_t)e * 3072, nullptr,
               768, 1, row_base, blockIdx.x * 64, As, Bs);
}

__global__ void __launch_bounds__(256) moe_ffn2_kernel(
    const float* hidden, const float* w2, const float* b2, float* y,
    const int* expert_of_slot)
{
    int row_base = blockIdx.y * 128;
    int e = expert_of_slot[row_base];
    if (e < 0) return;
    __shared__ bf16_t As[128][TPAD];
    __shared__ bf16_t Bs[64][TPAD];
    gemm_block(hidden, 3072, nullptr, w2 + (size_t)e * 3072 * 768, 768,
               y, 768, b2 + (size_t)e * 768, nullptr,
               3072, 0, row_base, blockIdx.x * 64, As, Bs);
}

// ---------------- Flash attention: 4 waves/block, 16 q-rows per wave, WMMA QK^T and AV ------
__global__ void __launch_bounds__(128) attn_kernel(const float* __restrict__ qkv, float* __restrict__ ao)
{
    const int bh = blockIdx.y;          // b*24 + h
    const int b  = bh / 24, h = bh % 24;
    const int tid  = threadIdx.x;
    const int lane = tid & 31;
    const int wv   = tid >> 5;          // 0..3
    const int g    = lane >> 4;
    const int lr   = lane & 15;

    __shared__ bf16_t Ks[32][TPAD];       // [key][d]
    __shared__ bf16_t Vt[32][TPAD];       // [d][key] (transposed for AV B-frag)
    __shared__ bf16_t Ps[4][16][TPAD];    // per-wave probs: [row][key]

    // Q fragment (16x32 A-matrix): lane row = l%16, element i -> d = (i<8?0:16)+8g+i%8
    const int qrow = blockIdx.x * 64 + wv * 16 + lr;
    const float* qp = qkv + (size_t)(b * 1600 + qrow) * 2304 + h * 32;
    float4 q0 = *(const float4*)(qp + g * 8);
    float4 q1 = *(const float4*)(qp + g * 8 + 4);
    float4 q2 = *(const float4*)(qp + 16 + g * 8);
    float4 q3 = *(const float4*)(qp + 16 + g * 8 + 4);
    v8bf qlo = __builtin_shufflevector(pack4(q0), pack4(q1), 0,1,2,3,4,5,6,7);
    v8bf qhi = __builtin_shufflevector(pack4(q2), pack4(q3), 0,1,2,3,4,5,6,7);
    v16bf qf = __builtin_shufflevector(qlo, qhi, 0,1,2,3,4,5,6,7,8,9,10,11,12,13,14,15);

    const float scale = 0.17677669529663687f; // 1/sqrt(32)
    float m_r[8], l_r[8];
    #pragma unroll
    for (int r = 0; r < 8; ++r) { m_r[r] = -1e30f; l_r[r] = 0.0f; }
    v8f o0 = {}, o1 = {};

    for (int kc = 0; kc < 1600; kc += 32) {
        __syncthreads();
        // stage K,V chunk (32 keys x 32 d) with b128 loads
        #pragma unroll
        for (int i = 0; i < 2; ++i) {
            int id  = i * 128 + tid;       // 256 quads
            int key = id >> 3, dq = id & 7;
            const float* kvp = qkv + (size_t)(b * 1600 + kc + key) * 2304 + h * 32 + dq * 4;
            float4 kq = *(const float4*)(kvp + 768);
            float4 vq = *(const float4*)(kvp + 1536);
            *(v4bf*)&Ks[key][dq * 4] = pack4(kq);
            v4bf vp = pack4(vq);
            Vt[dq * 4 + 0][key] = vp[0];
            Vt[dq * 4 + 1][key] = vp[1];
            Vt[dq * 4 + 2][key] = vp[2];
            Vt[dq * 4 + 3][key] = vp[3];
        }
        __syncthreads();

        // scores: Q[16x32] x K^T[32x16] per key-half
        v16bf kf0 = make_frag(&Ks[lr][g * 8],      &Ks[lr][16 + g * 8]);
        v16bf kf1 = make_frag(&Ks[16 + lr][g * 8], &Ks[16 + lr][16 + g * 8]);
        v8f s0 = {}, s1 = {};
        s0 = __builtin_amdgcn_wmma_f32_16x16x32_bf16(false, qf, false, kf0, (short)0, s0, false, false);
        s1 = __builtin_amdgcn_wmma_f32_16x16x32_bf16(false, qf, false, kf1, (short)0, s1, false, false);

        // online softmax per row (row r+8g lives across 16 lanes of this half-wave)
        #pragma unroll
        for (int r = 0; r < 8; ++r) {
            float a0 = s0[r] * scale, a1 = s1[r] * scale;
            float mv = fmaxf(a0, a1);
            #pragma unroll
            for (int o = 8; o > 0; o >>= 1) mv = fmaxf(mv, __shfl_xor(mv, o, 32));
            float mn = fmaxf(m_r[r], mv);
            float alpha = __expf(m_r[r] - mn);
            float p0 = __expf(a0 - mn), p1 = __expf(a1 - mn);
            float ps = p0 + p1;
            #pragma unroll
            for (int o = 8; o > 0; o >>= 1) ps += __shfl_xor(ps, o, 32);
            l_r[r] = l_r[r] * alpha + ps;
            m_r[r] = mn;
            o0[r] *= alpha; o1[r] *= alpha;
            int prow = r + 8 * g;
            Ps[wv][prow][lr]      = (bf16_t)p0;
            Ps[wv][prow][16 + lr] = (bf16_t)p1;
        }
        __syncthreads();

        // AV: P[16x32] x V[32x32]
        v16bf pf  = make_frag(&Ps[wv][lr][g * 8],  &Ps[wv][lr][16 + g * 8]);
        v16bf vf0 = make_frag(&Vt[lr][g * 8],      &Vt[lr][16 + g * 8]);
        v16bf vf1 = make_frag(&Vt[16 + lr][g * 8], &Vt[16 + lr][16 + g * 8]);
        o0 = __builtin_amdgcn_wmma_f32_16x16x32_bf16(false, pf, false, vf0, (short)0, o0, false, false);
        o1 = __builtin_amdgcn_wmma_f32_16x16x32_bf16(false, pf, false, vf1, (short)0, o1, false, false);
    }

    // write ao[b, m, h, d]
    #pragma unroll
    for (int r = 0; r < 8; ++r) {
        int m = blockIdx.x * 64 + wv * 16 + r + 8 * g;
        float inv = 1.0f / l_r[r];
        float* dst = ao + (size_t)(b * 1600 + m) * 768 + h * 32;
        dst[lr]      = o0[r] * inv;
        dst[16 + lr] = o1[r] * inv;
    }
}

// ---------------- LayerNorm ----------------
__global__ void __launch_bounds__(256) ln_kernel(
    const float* __restrict__ x, const float* __restrict__ gmm,
    const float* __restrict__ bta, float* __restrict__ out)
{
    const int row = blockIdx.x;
    const int tid = threadIdx.x;
    const float* xr = x + (size_t)row * 768;
    __shared__ float red[256];

    float s = 0.0f;
    for (int c = tid; c < 768; c += 256) s += xr[c];
    red[tid] = s; __syncthreads();
    for (int o = 128; o > 0; o >>= 1) { if (tid < o) red[tid] += red[tid + o]; __syncthreads(); }
    float mu = red[0] * (1.0f / 768.0f);
    __syncthreads();

    float v = 0.0f;
    for (int c = tid; c < 768; c += 256) { float d = xr[c] - mu; v += d * d; }
    red[tid] = v; __syncthreads();
    for (int o = 128; o > 0; o >>= 1) { if (tid < o) red[tid] += red[tid + o]; __syncthreads(); }
    float inv = rsqrtf(red[0] * (1.0f / 768.0f) + 1e-6f);

    float* outr = out + (size_t)row * 768;
    for (int c = tid; c < 768; c += 256)
        outr[c] = (xr[c] - mu) * inv * gmm[c] + bta[c];
}

// ---------------- Gating: wave per token, lane c<16 owns expert c ----------------
__global__ void __launch_bounds__(256) gate_kernel(
    const float* __restrict__ h2, const float* __restrict__ gw,
    int* __restrict__ topi, float* __restrict__ topw,
    float* mean_prob, float* topk_cnt, int* counts)
{
    const int wv = threadIdx.x >> 5, lane = threadIdx.x & 31;
    const int t = blockIdx.x * 8 + wv;
    if (t >= 3200) return;
    const int c = lane & 15, half = lane >> 4;

    const float* hr = h2 + (size_t)t * 768;
    float acc = 0.0f;
    for (int j = 0; j < 384; ++j) {
        int r = half * 384 + j;
        acc += hr[r] * gw[r * 16 + c];
    }
    acc += __shfl_xor(acc, 16, 32);   // combine halves -> full logit in all 32 lanes

    float mx = acc;
    #pragma unroll
    for (int o = 8; o > 0; o >>= 1) mx = fmaxf(mx, __shfl_xor(mx, o, 32));
    float p = __expf(acc - mx);
    float sum = p;
    #pragma unroll
    for (int o = 8; o > 0; o >>= 1) sum += __shfl_xor(sum, o, 32);
    p /= sum;

    // top-1 (lowest index wins ties, matching top_k)
    float v1 = p; int i1 = c;
    #pragma unroll
    for (int o = 8; o > 0; o >>= 1) {
        float ov = __shfl_xor(v1, o, 32); int oi = __shfl_xor(i1, o, 32);
        if (ov > v1 || (ov == v1 && oi < i1)) { v1 = ov; i1 = oi; }
    }
    float pv = (c == i1) ? -1.0f : p;
    float v2 = pv; int i2 = c;
    #pragma unroll
    for (int o = 8; o > 0; o >>= 1) {
        float ov = __shfl_xor(v2, o, 32); int oi = __shfl_xor(i2, o, 32);
        if (ov > v2 || (ov == v2 && oi < i2)) { v2 = ov; i2 = oi; }
    }
    float wsum = v1 + v2;

    if (lane == 0) {
        topi[t * 2 + 0] = i1; topi[t * 2 + 1] = i2;
        topw[t * 2 + 0] = v1 / wsum; topw[t * 2 + 1] = v2 / wsum;
        atomicAdd(&counts[i1], 1); atomicAdd(&counts[i2], 1);
    }
    if (lane < 16) {
        atomicAdd(&mean_prob[c], p * (1.0f / 3200.0f));
        atomicAdd(&topk_cnt[c], (p > v2 ? 1.0f : 0.0f) * (1.0f / 3200.0f));
    }
}

// ---------------- Routing helpers ----------------
__global__ void init_kernel(int* token_of_slot, int* expert_of_slot,
                            float* mean_prob, float* topk_cnt, int* counts)
{
    int t = blockIdx.x * 256 + threadIdx.x;
    if (t < MAXSLOTS) { token_of_slot[t] = -1; expert_of_slot[t] = -1; }
    if (t < 16) { mean_prob[t] = 0.0f; topk_cnt[t] = 0.0f; counts[t] = 0; }
}

__global__ void bases_kernel(const int* counts, int* base, int* cursor)
{
    if (threadIdx.x == 0 && blockIdx.x == 0) {
        int run = 0;
        for (int e = 0; e < 16; ++e) {
            base[e] = run;
            run += ((counts[e] + 127) >> 7) << 7;   // 128-align => single-expert GEMM tiles
            cursor[e] = 0;
        }
    }
}

__global__ void assign_kernel(const int* topi, const int* base, int* cursor,
                              int* token_of_slot, int* expert_of_slot, int* slot_of_token)
{
    int t = blockIdx.x * 256 + threadIdx.x;
    if (t >= 3200) return;
    for (int r = 0; r < 2; ++r) {
        int e = topi[t * 2 + r];
        int pos = atomicAdd(&cursor[e], 1);
        int slot = base[e] + pos;
        token_of_slot[slot] = t;
        expert_of_slot[slot] = e;
        slot_of_token[t * 2 + r] = slot;
    }
}

__global__ void combine_kernel(float* __restrict__ out, const float* __restrict__ y,
                               const int* __restrict__ slot_of_token, const float* __restrict__ topw)
{
    int idx = blockIdx.x * 256 + threadIdx.x;
    if (idx >= 3200 * 768) return;
    int t = idx / 768, c = idx % 768;
    int s0 = slot_of_token[t * 2], s1 = slot_of_token[t * 2 + 1];
    out[idx] += topw[t * 2] * y[(size_t)s0 * 768 + c] + topw[t * 2 + 1] * y[(size_t)s1 * 768 + c];
}

__global__ void aux_kernel(const float* mean_prob, const float* topk_cnt, float* out_scalar)
{
    if (threadIdx.x == 0 && blockIdx.x == 0) {
        float s = 0.0f;
        for (int e = 0; e < 16; ++e) s += mean_prob[e] * topk_cnt[e];
        *out_scalar = s * 16.0f;
    }
}

// ---------------- Host-side launcher ----------------
extern "C" void kernel_launch(void* const* d_in, const int* in_sizes, int n_in,
                              void* d_out, int out_size, void* d_ws, size_t ws_size,
                              hipStream_t stream)
{
    const float* x      = (const float*)d_in[0];
    const float* ln1_g  = (const float*)d_in[1];
    const float* ln1_b  = (const float*)d_in[2];
    const float* qkv_w  = (const float*)d_in[3];
    const float* proj_w = (const float*)d_in[4];
    const float* proj_b = (const float*)d_in[5];
    const float* ln2_g  = (const float*)d_in[6];
    const float* ln2_b  = (const float*)d_in[7];
    const float* gate_w = (const float*)d_in[8];
    const float* w1     = (const float*)d_in[9];
    const float* b1     = (const float*)d_in[10];
    const float* w2     = (const float*)d_in[11];
    const float* b2     = (const float*)d_in[12];
    float* out = (float*)d_out;                       // [2*1600*768] x, then [1] aux

    // workspace layout
    char* w = (char*)d_ws;
    float* h      = (float*)w; w += (size_t)3200 * 768  * 4;
    float* qkvb   = (float*)w; w += (size_t)3200 * 2304 * 4;
    float* ao     = (float*)w; w += (size_t)3200 * 768  * 4;
    float* h2     = (float*)w; w += (size_t)3200 * 768  * 4;
    float* hidden = (float*)w; w += (size_t)MAXSLOTS * 3072 * 4;
    float* yb     = (float*)w; w += (size_t)MAXSLOTS * 768  * 4;
    float* mean_prob = (float*)w; w += 16 * 4;
    float* topk_cnt  = (float*)w; w += 16 * 4;
    float* topw      = (float*)w; w += 6400 * 4;
    int* counts         = (int*)w; w += 16 * 4;
    int* basev          = (int*)w; w += 16 * 4;
    int* cursor         = (int*)w; w += 16 * 4;
    int* topi           = (int*)w; w += 6400 * 4;
    int* slot_of_token  = (int*)w; w += 6400 * 4;
    int* token_of_slot  = (int*)w; w += MAXSLOTS * 4;
    int* expert_of_slot = (int*)w; w += MAXSLOTS * 4;

    init_kernel<<<(MAXSLOTS + 255) / 256, 256, 0, stream>>>(
        token_of_slot, expert_of_slot, mean_prob, topk_cnt, counts);

    ln_kernel<<<3200, 256, 0, stream>>>(x, ln1_g, ln1_b, h);

    // qkv = h @ qkv_w  (3200 x 768 x 2304)
    gemm_kernel<<<dim3(2304 / 64, 3200 / 128), 256, 0, stream>>>(
        h, 768, qkv_w, 2304, qkvb, 2304, nullptr, nullptr, 768, 0);

    // attention over qkv
    attn_kernel<<<dim3(1600 / 64, 2 * 24), 128, 0, stream>>>(qkvb, ao);

    // out = x + ao @ proj_w + proj_b  (3200 x 768 x 768)
    gemm_kernel<<<dim3(768 / 64, 3200 / 128), 256, 0, stream>>>(
        ao, 768, proj_w, 768, out, 768, proj_b, x, 768, 0);

    ln_kernel<<<3200, 256, 0, stream>>>(out, ln2_g, ln2_b, h2);

    gate_kernel<<<3200 / 8, 256, 0, stream>>>(
        h2, gate_w, topi, topw, mean_prob, topk_cnt, counts);

    bases_kernel<<<1, 32, 0, stream>>>(counts, basev, cursor);

    assign_kernel<<<(3200 + 255) / 256, 256, 0, stream>>>(
        topi, basev, cursor, token_of_slot, expert_of_slot, slot_of_token);

    // hidden[slot] = relu(h2[token] @ w1[e] + b1[e])   (slots x 768 x 3072)
    moe_ffn1_kernel<<<dim3(3072 / 64, MAXSLOTS / 128), 256, 0, stream>>>(
        h2, w1, b1, hidden, token_of_slot, expert_of_slot);

    // y[slot] = hidden[slot] @ w2[e] + b2[e]           (slots x 3072 x 768)
    moe_ffn2_kernel<<<dim3(768 / 64, MAXSLOTS / 128), 256, 0, stream>>>(
        hidden, w2, b2, yb, expert_of_slot);

    combine_kernel<<<(3200 * 768) / 256, 256, 0, stream>>>(out, yb, slot_of_token, topw);

    aux_kernel<<<1, 32, 0, stream>>>(mean_prob, topk_cnt, out + (size_t)2 * 1600 * 768);

    (void)in_sizes; (void)n_in; (void)out_size; (void)ws_size;
}
